// StockPredictionModel_41540923687151
// MI455X (gfx1250) — matrix operand order repaired and checked by reading, not compiled, not run on hardware
//
#include <hip/hip_runtime.h>
#include <hip/hip_bf16.h>
#include <math.h>

// ---------------------------------------------------------------------------
// Model dims (fixed by the reference)
// ---------------------------------------------------------------------------
#define NS   4096   // stocks
#define SEQ  64     // sequence length
#define FIN  64     // input features
#define DH0  128    // GRU layer0 hidden
#define DH1  64     // GRU layer1 hidden
#define FD   64     // feature dim (MHCA)
#define NLAT 32     // latent rows

typedef __attribute__((ext_vector_type(16))) _Float16 v16h;
typedef __attribute__((ext_vector_type(8)))  float    v8f;

// ---------------------------------------------------------------------------
// WMMA helpers (wave32, CDNA5 16x16x32 f16 -> f32)
// ---------------------------------------------------------------------------
__device__ __forceinline__ v8f wmma32(v16h a, v16h b, v8f c) {
  return __builtin_amdgcn_wmma_f32_16x16x32_f16(false, a, false, b, (short)0, c,
                                                false, false);
}

// Fragment layouts (wave32):
//  A (16x32 f16): lane 0-15 row=lane, K = h + 8*(h>=8); lane 16-31 row=lane-16, +8
//  B (32x16 f16): lane col = lane&15, K = k0 + a_kmap(h, lane), same order as A
//  C/D (16x16 f32): lane col = lane&15, row = j + 8*(lane>=16)
// Packed buffers store one 32-byte (16-half) fragment slice per lane,
// contiguous, so a fragment load is a single aligned v16h load (2x b128).
__device__ __forceinline__ v16h ldfragG(const _Float16* p, long idx) {
  return *(const v16h*)(p + (idx * 32 + (threadIdx.x & 31)) * 16);
}
__device__ __forceinline__ v16h ldfragS(const _Float16* p) {
  return *(const v16h*)(p + (size_t)(threadIdx.x & 31) * 16);
}

// K-position inside a 32-chunk for A-fragment half h / lane group
__device__ __forceinline__ int a_kmap(int h, int lane) {
  return h + ((h >> 3) << 3) + ((lane >> 4) << 3);
}
// inverse: local K (0..31) -> (lane-group hi, half index h)
__device__ __forceinline__ void a_slot(int kl, int& hi, int& h) {
  if (kl < 8)       { hi = 0; h = kl; }
  else if (kl < 16) { hi = 1; h = kl - 8; }
  else if (kl < 24) { hi = 0; h = kl - 8; }
  else              { hi = 1; h = kl - 16; }
}

__device__ __forceinline__ float sigmoidf_(float x) {
  return 1.0f / (1.0f + __expf(-x));
}

// order-preserving float<->int map for atomicMax-based segment max
__device__ __forceinline__ int f2ord(float f) {
  int i = __float_as_int(f);
  return i >= 0 ? i : (i ^ 0x7fffffff);
}
__device__ __forceinline__ float ord2f(int i) {
  return __int_as_float(i >= 0 ? i : (i ^ 0x7fffffff));
}

// ---------------------------------------------------------------------------
// Pack kernels: one-time f32 -> f16 swizzle into WMMA fragment layouts
// ---------------------------------------------------------------------------
// B pack: W (K x N, row-major f32) -> out[((kc*ntn + nt)*32 + lane)*16 + h]
__global__ void pack_b_kernel(const float* __restrict__ W,
                              _Float16* __restrict__ out, int K, int N,
                              int ntn) {
  long g = blockIdx.x * (long)blockDim.x + threadIdx.x;
  long tot = (long)(K >> 5) * ntn * 512;
  if (g >= tot) return;
  int h = (int)(g & 15);
  int lane = (int)((g >> 4) & 31);
  long rest = g >> 9;
  int nt = (int)(rest % ntn);
  int kc = (int)(rest / ntn);
  int col = (nt << 4) + (lane & 15);
  int k = (kc << 5) + a_kmap(h, lane);
  float v = (col < N) ? W[(size_t)k * N + col] : 0.0f;
  out[g] = (_Float16)v;
}

// A pack: A (M x K, row stride lda) -> out[((mt*nkc + kc)*32 + lane)*16 + h]
__global__ void pack_a_kernel(const float* __restrict__ A, long lda,
                              _Float16* __restrict__ out, int M, int K) {
  long g = blockIdx.x * (long)blockDim.x + threadIdx.x;
  int nkc = K >> 5;
  long tot = (long)(M >> 4) * nkc * 512;
  if (g >= tot) return;
  int h = (int)(g & 15);
  int lane = (int)((g >> 4) & 31);
  long rest = g >> 9;
  int kc = (int)(rest % nkc);
  int mt = (int)(rest / nkc);
  int row = (mt << 4) + (lane & 15);
  int k = (kc << 5) + a_kmap(h, lane);
  out[g] = (_Float16)A[(size_t)row * lda + k];
}

// x_seq pack: (NS x SEQ x FIN) -> out[(((rt*SEQ + t)*2 + kc)*32 + lane)*16 + h]
__global__ void pack_x_kernel(const float* __restrict__ x,
                              _Float16* __restrict__ out) {
  long g = blockIdx.x * (long)blockDim.x + threadIdx.x;
  long tot = (long)(NS >> 4) * SEQ * 2 * 512;
  if (g >= tot) return;
  int h = (int)(g & 15);
  int lane = (int)((g >> 4) & 31);
  long rest = g >> 9;
  int kc = (int)(rest & 1);
  long r2 = rest >> 1;
  int t = (int)(r2 & (SEQ - 1));
  int rt = (int)(r2 >> 6);
  int row = (rt << 4) + (lane & 15);
  int k = (kc << 5) + a_kmap(h, lane);
  out[g] = (_Float16)x[(size_t)row * (SEQ * FIN) + t * FIN + k];
}

// ---------------------------------------------------------------------------
// Packed GRU weight blob: all 14 matrices contiguous (offsets in halves).
// Sizes: 64x128 = 8192, 128x128 = 16384, 128x64 = 8192, 64x64 = 4096 halves.
// ---------------------------------------------------------------------------
#define O_WZ0 0
#define O_WK0 8192
#define O_WV0 16384
#define O_WH0 24576
#define O_UZ0 32768
#define O_WQ0 49152
#define O_UH0 65536
#define O_WZ1 81920
#define O_WK1 90112
#define O_WV1 98304
#define O_WH1 106496
#define O_UZ1 114688
#define O_WQ1 118784
#define O_UH1 122880
#define WL_TOT 126976  // total halves (= 248 KB)

// ---------------------------------------------------------------------------
// Fused 2-layer attention-GRU scan. One block = one 16-row tile, 8 waves.
// All weights staged in LDS (f16 fragment layout) once; h0/h1 also live in
// LDS in packed A-fragment f16 layout across all 64 timesteps.
// ---------------------------------------------------------------------------
struct GruParams {
  const _Float16* xp;     // packed x_seq A-fragments
  const _Float16* wpack;  // all 14 weight matrices, packed B-fragments
  float* A1;              // NS x DH1 output (final h1)
  const float *bWz0, *bUz0, *bWq0, *bWk0, *bWv0, *bWh0, *bUh0;
  const float *bWz1, *bUz1, *bWq1, *bWk1, *bWv1, *bWh1, *bUh1;
};

__global__ __launch_bounds__(256) void fused_gru_kernel(GruParams P) {
  __shared__ __align__(32) _Float16 wl[WL_TOT];    // 248 KB weights
  __shared__ __align__(32) _Float16 h0p[4 * 512];  // 4 K-chunks of h0
  __shared__ __align__(32) _Float16 h1p[2 * 512];  // 2 K-chunks of h1
  __shared__ float rs0[16];
  __shared__ float rs1[16];

  const int tid = threadIdx.x;
  const int wave = tid >> 5;
  const int lane = tid & 31;
  const int row0 = blockIdx.x << 4;

  // stage all packed weights into LDS (16B vector copy)
  {
    const uint4* s = (const uint4*)P.wpack;
    uint4* d = (uint4*)wl;
    for (int i = tid; i < WL_TOT * 2 / 16; i += 256) d[i] = s[i];
  }
  for (int i = tid; i < 4 * 512; i += 256) h0p[i] = (_Float16)0.0f;
  for (int i = tid; i < 2 * 512; i += 256) h1p[i] = (_Float16)0.0f;
  __syncthreads();

  const int colL = lane & 15;
  const int rbase = (lane >> 4) << 3;
  const int c0 = (wave << 4) + colL;        // layer0 col (0..127), 8 waves
  const int c1 = ((wave & 3) << 4) + colL;  // layer1 col (0..63), waves 0-3

  // LDS write slot for this thread's h0/h1 column (row r -> base + r*16)
  int hi0, hh0, hi1, hh1;
  a_slot(c0 & 31, hi0, hh0);
  a_slot(c1 & 31, hi1, hh1);
  const int slot0 = ((c0 >> 5) * 32 + rbase + 16 * hi0) * 16 + hh0;
  const int slot1 = ((c1 >> 5) * 32 + rbase + 16 * hi1) * 16 + hh1;

  const float bz0 = P.bWz0[c0] + P.bUz0[c0];
  const float bq0 = P.bWq0[c0], bk0 = P.bWk0[c0], bv0 = P.bWv0[c0];
  const float bh0 = P.bWh0[c0], bu0 = P.bUh0[c0];
  const float bz1 = P.bWz1[c1] + P.bUz1[c1];
  const float bq1 = P.bWq1[c1], bk1 = P.bWk1[c1], bv1 = P.bWv1[c1];
  const float bh1 = P.bWh1[c1], bu1 = P.bUh1[c1];

  const long xbase = (long)blockIdx.x * SEQ * 2;  // x fragment index base

  for (int t = 0; t < SEQ; ++t) {
    // ---------------- layer 0 (din=64 -> dh=128) ----------------
    v16h ax[2];
    ax[0] = ldfragG(P.xp, xbase + t * 2 + 0);
    ax[1] = ldfragG(P.xp, xbase + t * 2 + 1);
    if (t + 1 < SEQ)  // prefetch next timestep's x fragments
      __builtin_prefetch(P.xp + ((xbase + (t + 1) * 2) * 32 + lane) * 16, 0, 1);
    v16h ah[4];
#pragma unroll
    for (int kc = 0; kc < 4; ++kc) ah[kc] = ldfragS(&h0p[kc * 512]);

    v8f zA = {}, qA = {}, kA = {}, vA = {}, hA = {}, uA = {};
#pragma unroll
    for (int kc = 0; kc < 2; ++kc) {
      const int o = (kc * 8 + wave) * 512;  // ntn=8
      zA = wmma32(ax[kc], ldfragS(&wl[O_WZ0 + o]), zA);
      kA = wmma32(ax[kc], ldfragS(&wl[O_WK0 + o]), kA);
      vA = wmma32(ax[kc], ldfragS(&wl[O_WV0 + o]), vA);
      hA = wmma32(ax[kc], ldfragS(&wl[O_WH0 + o]), hA);
    }
#pragma unroll
    for (int kc = 0; kc < 4; ++kc) {
      const int o = (kc * 8 + wave) * 512;
      zA = wmma32(ah[kc], ldfragS(&wl[O_UZ0 + o]), zA);
      qA = wmma32(ah[kc], ldfragS(&wl[O_WQ0 + o]), qA);
      uA = wmma32(ah[kc], ldfragS(&wl[O_UH0 + o]), uA);
    }
    float dp[8];
#pragma unroll
    for (int j = 0; j < 8; ++j) dp[j] = (qA[j] + bq0) * (kA[j] + bk0);
#pragma unroll
    for (int m = 1; m < 16; m <<= 1)
#pragma unroll
      for (int j = 0; j < 8; ++j) dp[j] += __shfl_xor(dp[j], m, 32);

    __syncthreads();
    if (tid < 16) rs0[tid] = 0.0f;
    __syncthreads();
    if (colL == 0) {
#pragma unroll
      for (int j = 0; j < 8; ++j) atomicAdd(&rs0[rbase + j], dp[j]);
    }
    __syncthreads();
#pragma unroll
    for (int j = 0; j < 8; ++j) {
      const int r = rbase + j;
      const float a = sigmoidf_(rs0[r] * 0.08838834764831845f);  // 1/sqrt(128)
      const float zs = sigmoidf_(zA[j] + bz0);
      const float ht = tanhf(hA[j] + bh0 + a * (vA[j] + bv0) * (uA[j] + bu0));
      const float hold = (float)h0p[slot0 + j * 16];
      h0p[slot0 + j * 16] = (_Float16)((1.0f - zs) * hold + zs * ht);
    }
    __syncthreads();

    // ---------------- layer 1 (din=128 -> dh=64) ----------------
    v8f zB = {}, qB = {}, kB = {}, vB = {}, hB = {}, uB = {};
    float dp1[8];
    if (wave < 4) {
      v16h bx[4];
#pragma unroll
      for (int kc = 0; kc < 4; ++kc) bx[kc] = ldfragS(&h0p[kc * 512]);
      v16h bhf[2];
      bhf[0] = ldfragS(&h1p[0]);
      bhf[1] = ldfragS(&h1p[512]);
#pragma unroll
      for (int kc = 0; kc < 4; ++kc) {
        const int o = (kc * 4 + (wave & 3)) * 512;  // ntn=4
        zB = wmma32(bx[kc], ldfragS(&wl[O_WZ1 + o]), zB);
        kB = wmma32(bx[kc], ldfragS(&wl[O_WK1 + o]), kB);
        vB = wmma32(bx[kc], ldfragS(&wl[O_WV1 + o]), vB);
        hB = wmma32(bx[kc], ldfragS(&wl[O_WH1 + o]), hB);
      }
#pragma unroll
      for (int kc = 0; kc < 2; ++kc) {
        const int o = (kc * 4 + (wave & 3)) * 512;
        zB = wmma32(bhf[kc], ldfragS(&wl[O_UZ1 + o]), zB);
        qB = wmma32(bhf[kc], ldfragS(&wl[O_WQ1 + o]), qB);
        uB = wmma32(bhf[kc], ldfragS(&wl[O_UH1 + o]), uB);
      }
#pragma unroll
      for (int j = 0; j < 8; ++j) dp1[j] = (qB[j] + bq1) * (kB[j] + bk1);
#pragma unroll
      for (int m = 1; m < 16; m <<= 1)
#pragma unroll
        for (int j = 0; j < 8; ++j) dp1[j] += __shfl_xor(dp1[j], m, 32);
    }
    __syncthreads();
    if (tid < 16) rs1[tid] = 0.0f;
    __syncthreads();
    if (wave < 4 && colL == 0) {
#pragma unroll
      for (int j = 0; j < 8; ++j) atomicAdd(&rs1[rbase + j], dp1[j]);
    }
    __syncthreads();
    if (wave < 4) {
#pragma unroll
      for (int j = 0; j < 8; ++j) {
        const int r = rbase + j;
        const float a = sigmoidf_(rs1[r] * 0.125f);  // 1/sqrt(64)
        const float zs = sigmoidf_(zB[j] + bz1);
        const float ht = tanhf(hB[j] + bh1 + a * (vB[j] + bv1) * (uB[j] + bu1));
        const float hold = (float)h1p[slot1 + j * 16];
        const float hn = (1.0f - zs) * hold + zs * ht;
        h1p[slot1 + j * 16] = (_Float16)hn;
        if (t == SEQ - 1) P.A1[(size_t)(row0 + r) * DH1 + c1] = hn;
      }
    }
    __syncthreads();
  }
}

// ---------------------------------------------------------------------------
// Generic WMMA GEMM on packed fragments:
// C(MxN) = act(A @ B + bias). One wave = one 16x16 tile; 4 waves / block.
// ---------------------------------------------------------------------------
__global__ __launch_bounds__(128) void gemm_wmma_kernel(
    const _Float16* __restrict__ Ap, const _Float16* __restrict__ Bp,
    const float* __restrict__ bias, float* __restrict__ C, int M, int Ncols,
    int K, int act) {
  const int wave = threadIdx.x >> 5;
  const int lane = threadIdx.x & 31;
  const int ntn = (Ncols + 15) >> 4;
  const int nkc = K >> 5;
  const int total = (M >> 4) * ntn;
  const int tile = blockIdx.x * 4 + wave;
  if (tile >= total) return;
  const int mt = tile / ntn;
  const int nt = tile - mt * ntn;
  const int col = (nt << 4) + (lane & 15);
  v8f acc = {};
  for (int kc = 0; kc < nkc; ++kc)
    acc = wmma32(ldfragG(Ap, (long)mt * nkc + kc),
                 ldfragG(Bp, (long)kc * ntn + nt), acc);
  if (col < Ncols) {
    const float bb = bias ? bias[col] : 0.0f;
    const int r0 = (mt << 4) + ((lane >> 4) << 3);
#pragma unroll
    for (int j = 0; j < 8; ++j) {
      float v = acc[j] + bb;
      if (act == 1) v = v > 0.0f ? v : (__expf(v) - 1.0f);  // ELU
      C[(size_t)(r0 + j) * Ncols + col] = v;
    }
  }
}

// ---------------------------------------------------------------------------
// Small utility kernels
// ---------------------------------------------------------------------------
__global__ void fill_f32_kernel(float* p, float v, long n) {
  long g = blockIdx.x * (long)blockDim.x + threadIdx.x;
  if (g < n) p[g] = v;
}
__global__ void fill_i32_kernel(int* p, int v, long n) {
  long g = blockIdx.x * (long)blockDim.x + threadIdx.x;
  if (g < n) p[g] = v;
}
__global__ void reduce_sum_kernel(const float* x, int n, float* out) {
  __shared__ float sb[256];
  float s = 0.0f;
  for (int i = blockIdx.x * blockDim.x + threadIdx.x; i < n;
       i += gridDim.x * blockDim.x)
    s += x[i];
  sb[threadIdx.x] = s;
  __syncthreads();
  for (int k = 128; k > 0; k >>= 1) {
    if (threadIdx.x < k) sb[threadIdx.x] += sb[threadIdx.x + k];
    __syncthreads();
  }
  if (threadIdx.x == 0) atomicAdd(out, sb[0]);
}

// ---------------------------------------------------------------------------
// GAT kernels
// ---------------------------------------------------------------------------
__global__ void node_scores_kernel(const float* xh, const float* att_src,
                                   const float* att_dst, float* a_src,
                                   float* a_dst, int Nn, int heads, int oc) {
  int g = blockIdx.x * blockDim.x + threadIdx.x;
  if (g >= Nn * heads) return;
  int n = g / heads, h = g - n * heads;
  const float* xr = xh + ((size_t)n * heads + h) * oc;
  const float* as = att_src + h * oc;
  const float* ad = att_dst + h * oc;
  float s0 = 0.0f, s1 = 0.0f;
  for (int c = 0; c < oc; ++c) {
    s0 += xr[c] * as[c];
    s1 += xr[c] * ad[c];
  }
  a_src[g] = s0;
  a_dst[g] = s1;
}

__global__ void compute_ce_kernel(const float* wedge, const float* attedge,
                                  float* ce, int heads, int oc) {
  int h = threadIdx.x;
  if (h >= heads) return;
  float s = 0.0f;
  for (int c = 0; c < oc; ++c) s += wedge[h * oc + c] * attedge[h * oc + c];
  ce[h] = s;
}

__global__ void edge_alpha_max_kernel(const int* src, const int* dst,
                                      const float* ew, const float* ewsum,
                                      const float* a_src, const float* a_dst,
                                      const float* ce, float* alpha, int* mbits,
                                      int E, int Nn, int heads) {
  int e = blockIdx.x * blockDim.x + threadIdx.x;
  int Etot = E + Nn;
  if (e >= Etot) return;
  int s_, d_;
  float w;
  if (e < E) {
    s_ = src[e];
    d_ = dst[e];
    w = ew[e];
  } else {
    s_ = d_ = e - E;
    w = ewsum[0] / (float)E;  // mean edge attr for self-loops
  }
  for (int h = 0; h < heads; ++h) {
    float al = a_src[s_ * heads + h] + a_dst[d_ * heads + h] + w * ce[h];
    al = al > 0.0f ? al : 0.2f * al;  // leaky relu
    alpha[(size_t)e * heads + h] = al;
    atomicMax(&mbits[d_ * heads + h], f2ord(al));
  }
}

__global__ void edge_exp_sum_kernel(const int* dst, const float* alpha,
                                    const int* mbits, float* ex, float* denom,
                                    int E, int Nn, int heads) {
  int e = blockIdx.x * blockDim.x + threadIdx.x;
  int Etot = E + Nn;
  if (e >= Etot) return;
  int d_ = (e < E) ? dst[e] : (e - E);
  for (int h = 0; h < heads; ++h) {
    float v = __expf(alpha[(size_t)e * heads + h] - ord2f(mbits[d_ * heads + h]));
    ex[(size_t)e * heads + h] = v;
    atomicAdd(&denom[d_ * heads + h], v);
  }
}

__global__ void edge_scatter_kernel(const int* src, const int* dst,
                                    const float* ex, const float* denom,
                                    const float* xh, float* acc, int E, int Nn,
                                    int heads, int oc) {
  long g = blockIdx.x * (long)blockDim.x + threadIdx.x;
  long tot = (long)(E + Nn) * heads;
  if (g >= tot) return;
  int e = (int)(g / heads), h = (int)(g - (long)e * heads);
  int s_, d_;
  if (e < E) {
    s_ = src[e];
    d_ = dst[e];
  } else {
    s_ = d_ = e - E;
  }
  float w = ex[(size_t)e * heads + h] / (denom[d_ * heads + h] + 1e-16f);
  const float* xr = xh + ((size_t)s_ * heads + h) * oc;
  float* ar = acc + ((size_t)d_ * heads + h) * oc;
  __builtin_prefetch(xr, 0, 1);  // global_prefetch_b8 on gathered source row
  for (int c = 0; c < oc; ++c) atomicAdd(&ar[c], w * xr[c]);
}

__global__ void gat_finalize_kernel(const float* acc, const float* bias,
                                    float* out, int Nn, int heads, int oc,
                                    int concat, int elu) {
  int g = blockIdx.x * blockDim.x + threadIdx.x;
  if (concat) {
    if (g >= Nn * heads * oc) return;
    int col = g % (heads * oc);
    float v = acc[g] + bias[col];
    if (elu) v = v > 0.0f ? v : (__expf(v) - 1.0f);
    out[g] = v;
  } else {
    if (g >= Nn * oc) return;
    int n = g / oc, c = g - n * oc;
    float s = 0.0f;
    for (int h = 0; h < heads; ++h) s += acc[((size_t)n * heads + h) * oc + c];
    float v = s / (float)heads + bias[c];
    if (elu) v = v > 0.0f ? v : (__expf(v) - 1.0f);
    out[g] = v;
  }
}

// ---------------------------------------------------------------------------
// MHCA attention core (heads=4, hd=16, L=32): K/V staged in LDS
// ---------------------------------------------------------------------------
__global__ __launch_bounds__(256) void mhca_attn_kernel(const float* Q,
                                                        const float* K,
                                                        const float* V,
                                                        float* O, int Nn) {
  __shared__ float sk[NLAT * FD];
  __shared__ float sv[NLAT * FD];
  for (int i = threadIdx.x; i < NLAT * FD; i += 256) {
    sk[i] = K[i];
    sv[i] = V[i];
  }
  __syncthreads();
  int g = blockIdx.x * 256 + threadIdx.x;
  if (g >= Nn * 4) return;
  const int n = g >> 2, h = g & 3;
  const float* q = Q + (size_t)n * FD + h * 16;
  float s[NLAT];
  float mx = -1e30f;
  for (int l = 0; l < NLAT; ++l) {
    float d = 0.0f;
    for (int k = 0; k < 16; ++k) d += q[k] * sk[l * FD + h * 16 + k];
    d *= 0.25f;  // 1/sqrt(16)
    s[l] = d;
    mx = fmaxf(mx, d);
  }
  float den = 0.0f;
  for (int l = 0; l < NLAT; ++l) {
    s[l] = __expf(s[l] - mx);
    den += s[l];
  }
  const float inv = 1.0f / den;
  float o[16];
#pragma unroll
  for (int k = 0; k < 16; ++k) o[k] = 0.0f;
  for (int l = 0; l < NLAT; ++l) {
    const float w = s[l];
    for (int k = 0; k < 16; ++k) o[k] += w * sv[l * FD + h * 16 + k];
  }
  for (int k = 0; k < 16; ++k) O[(size_t)n * FD + h * 16 + k] = o[k] * inv;
}

__global__ void concat_kernel(const float* B1, const float* B2, float* C,
                              int Nn) {
  int g = blockIdx.x * blockDim.x + threadIdx.x;
  if (g >= Nn * 128) return;
  int n = g >> 7, j = g & 127;
  C[g] = (j < 64) ? B1[n * 64 + j] : B2[n * 64 + (j - 64)];
}

// ---------------------------------------------------------------------------
// Host orchestration
// ---------------------------------------------------------------------------
static void pack_b(hipStream_t st, const float* W, int K, int N, _Float16* out) {
  const int ntn = (N + 15) / 16;
  const long tot = (long)(K / 32) * ntn * 512;
  pack_b_kernel<<<(int)((tot + 255) / 256), 256, 0, st>>>(W, out, K, N, ntn);
}

// packed A/B scratch + gemm in one call
static void launch_gemm(hipStream_t st, const float* A, long lda, const float* B,
                        const float* bias, float* C, int M, int Ncols, int K,
                        int act, _Float16* apack, _Float16* bpack) {
  const long atot = (long)(M / 16) * (K / 32) * 512;
  pack_a_kernel<<<(int)((atot + 255) / 256), 256, 0, st>>>(A, lda, apack, M, K);
  pack_b(st, B, K, Ncols, bpack);
  const int ntn = (Ncols + 15) / 16;
  const int total = (M / 16) * ntn;
  gemm_wmma_kernel<<<(total + 3) / 4, 128, 0, st>>>(apack, bpack, bias, C, M,
                                                    Ncols, K, act);
}

struct GatBufs {
  float *xh, *asrc, *adst, *ce, *alpha, *ex, *denom, *acc;
  int* mbits;
  _Float16 *apack, *bpack;
};

static void run_gat(hipStream_t st, const float* X, long lda, int din, int heads,
                    int oc, int concat, int elu, const float* w,
                    const float* attdst, const float* attedge,
                    const float* attsrc, const float* bias, const float* wedge,
                    const int* esrc, const int* edst, const float* ew,
                    const float* ewsum, int E, const GatBufs& B, float* out) {
  const int Nn = NS;
  launch_gemm(st, X, lda, w, nullptr, B.xh, Nn, heads * oc, din, 0, B.apack,
              B.bpack);
  node_scores_kernel<<<(Nn * heads + 255) / 256, 256, 0, st>>>(
      B.xh, attsrc, attdst, B.asrc, B.adst, Nn, heads, oc);
  compute_ce_kernel<<<1, 32, 0, st>>>(wedge, attedge, B.ce, heads, oc);
  fill_i32_kernel<<<(Nn * heads + 255) / 256, 256, 0, st>>>(
      B.mbits, (int)0x807fffff /* ord(-inf) */, Nn * heads);
  fill_f32_kernel<<<(Nn * heads + 255) / 256, 256, 0, st>>>(B.denom, 0.0f,
                                                            Nn * heads);
  fill_f32_kernel<<<((long)Nn * heads * oc + 255) / 256, 256, 0, st>>>(
      B.acc, 0.0f, (long)Nn * heads * oc);
  const int Etot = E + Nn;
  edge_alpha_max_kernel<<<(Etot + 255) / 256, 256, 0, st>>>(
      esrc, edst, ew, ewsum, B.asrc, B.adst, B.ce, B.alpha, B.mbits, E, Nn,
      heads);
  edge_exp_sum_kernel<<<(Etot + 255) / 256, 256, 0, st>>>(edst, B.alpha, B.mbits,
                                                          B.ex, B.denom, E, Nn,
                                                          heads);
  edge_scatter_kernel<<<(int)(((long)Etot * heads + 255) / 256), 256, 0, st>>>(
      esrc, edst, B.ex, B.denom, B.xh, B.acc, E, Nn, heads, oc);
  const int tot = concat ? Nn * heads * oc : Nn * oc;
  gat_finalize_kernel<<<(tot + 255) / 256, 256, 0, st>>>(B.acc, bias, out, Nn,
                                                         heads, oc, concat, elu);
}

static void run_mhca(hipStream_t st, const float* Xin, const float* R,
                     const float* Qw, const float* Qb, const float* Kw,
                     const float* Kb, const float* Vw, const float* Vb,
                     const float* Ow, const float* Ob, float* qbuf, float* kbuf,
                     float* vbuf, float* attn, float* out, _Float16* apack,
                     _Float16* bpack) {
  launch_gemm(st, Xin, FD, Qw, Qb, qbuf, NS, FD, FD, 0, apack, bpack);
  launch_gemm(st, R, FD, Kw, Kb, kbuf, NLAT, FD, FD, 0, apack, bpack);
  launch_gemm(st, R, FD, Vw, Vb, vbuf, NLAT, FD, FD, 0, apack, bpack);
  mhca_attn_kernel<<<(NS * 4 + 255) / 256, 256, 0, st>>>(qbuf, kbuf, vbuf, attn,
                                                         NS);
  launch_gemm(st, attn, FD, Ow, Ob, out, NS, FD, FD, 0, apack, bpack);
}

// Input leaf order: jax pytree (dicts sorted by key).
// 0:x_seq 1:edge_index 2:edge_weight 3:R1 4:R2
// att1(5..12): K.b K.w O.b O.w Q.b Q.w V.b V.w ; att2(13..20) same
// gat1a(21..26), gat1b(27..32), gat2a(33..38), gat2b(39..44):
//   att_dst att_edge att_src bias w w_edge
// gru0(45..58), gru1(59..72): U_h.b U_h.w U_z.b U_z.w W_h.b W_h.w
//   W_k.b W_k.w W_q.b W_q.w W_v.b W_v.w W_z.b W_z.w
extern "C" void kernel_launch(void* const* d_in, const int* in_sizes, int n_in,
                              void* d_out, int out_size, void* d_ws,
                              size_t ws_size, hipStream_t stream) {
  (void)n_in;
  (void)out_size;
  (void)ws_size;
  const float* x_seq = (const float*)d_in[0];
  const int* eidx = (const int*)d_in[1];
  const float* ew = (const float*)d_in[2];
  const int NE = in_sizes[1] / 2;
  const int* esrc = eidx;
  const int* edst = eidx + NE;
  const float** in = (const float**)d_in;

  // workspace bump allocator
  size_t off = 0;
  auto allocb = [&](size_t bytes) -> void* {
    void* p = (char*)d_ws + off;
    off += (bytes + 255) & ~(size_t)255;
    return p;
  };
  auto allocf = [&](size_t elems) -> float* {
    return (float*)allocb(elems * sizeof(float));
  };
  auto alloch = [&](size_t elems) -> _Float16* {
    return (_Float16*)allocb(elems * sizeof(_Float16));
  };

  float* A1 = allocf((size_t)NS * DH1);
  float* g1 = allocf((size_t)NS * 256);
  float* A2 = allocf((size_t)NS * FD);
  float* B1 = allocf((size_t)NS * FD);
  float* B2 = allocf((size_t)NS * FD);
  float* cbuf = allocf((size_t)NS * 128);
  float* g2 = allocf((size_t)NS * 256);
  float* qbuf = allocf((size_t)NS * FD);
  float* kbuf = allocf((size_t)NLAT * FD);
  float* vbuf = allocf((size_t)NLAT * FD);
  float* attn = allocf((size_t)NS * FD);
  GatBufs GB;
  GB.xh = allocf((size_t)NS * 256);
  GB.asrc = allocf((size_t)NS * 4);
  GB.adst = allocf((size_t)NS * 4);
  GB.ce = allocf(8);
  GB.alpha = allocf((size_t)(NE + NS) * 4);
  GB.ex = allocf((size_t)(NE + NS) * 4);
  GB.mbits = (int*)allocf((size_t)NS * 4);
  GB.denom = allocf((size_t)NS * 4);
  GB.acc = allocf((size_t)NS * 256);
  GB.apack = alloch((size_t)NS * 256);   // max M*K halves
  GB.bpack = alloch((size_t)256 * 256);  // max K*ceil16(N)*16 halves
  float* ewsum = allocf(1);

  _Float16* xpack = alloch((size_t)NS * SEQ * FIN);
  _Float16* wpack = alloch((size_t)WL_TOT);  // all GRU weights, one blob

  // --- pack x_seq + GRU weights into WMMA fragment layouts ---
  {
    const long xt = (long)(NS / 16) * SEQ * 2 * 512;
    pack_x_kernel<<<(int)((xt + 255) / 256), 256, 0, stream>>>(x_seq, xpack);
  }
  pack_b(stream, in[58], 64, 128, wpack + O_WZ0);
  pack_b(stream, in[52], 64, 128, wpack + O_WK0);
  pack_b(stream, in[56], 64, 128, wpack + O_WV0);
  pack_b(stream, in[50], 64, 128, wpack + O_WH0);
  pack_b(stream, in[48], 128, 128, wpack + O_UZ0);
  pack_b(stream, in[54], 128, 128, wpack + O_WQ0);
  pack_b(stream, in[46], 128, 128, wpack + O_UH0);
  pack_b(stream, in[72], 128, 64, wpack + O_WZ1);
  pack_b(stream, in[66], 128, 64, wpack + O_WK1);
  pack_b(stream, in[70], 128, 64, wpack + O_WV1);
  pack_b(stream, in[64], 128, 64, wpack + O_WH1);
  pack_b(stream, in[62], 64, 64, wpack + O_UZ1);
  pack_b(stream, in[68], 64, 64, wpack + O_WQ1);
  pack_b(stream, in[60], 64, 64, wpack + O_UH1);

  // --- fused 2-layer attention-GRU over all 64 timesteps -> A1 (4096x64) ---
  GruParams P;
  P.xp = xpack;
  P.wpack = wpack;
  P.A1 = A1;
  P.bWz0 = in[57]; P.bUz0 = in[47]; P.bWq0 = in[53]; P.bWk0 = in[51];
  P.bWv0 = in[55]; P.bWh0 = in[49]; P.bUh0 = in[45];
  P.bWz1 = in[71]; P.bUz1 = in[61]; P.bWq1 = in[67]; P.bWk1 = in[65];
  P.bWv1 = in[69]; P.bWh1 = in[63]; P.bUh1 = in[59];
  fused_gru_kernel<<<NS / 16, 256, 0, stream>>>(P);

  // --- mean(edge_weight) for self-loop attrs ---
  fill_f32_kernel<<<1, 1, 0, stream>>>(ewsum, 0.0f, 1);
  reduce_sum_kernel<<<256, 256, 0, stream>>>(ew, NE, ewsum);

  // --- gat1a: x_seq[:, -1, :] -> 256, elu -> g1 ---
  run_gat(stream, x_seq + (SEQ - 1) * FIN, (long)SEQ * FIN, FIN, 4, 64, 1, 1,
          in[25], in[21], in[22], in[23], in[24], in[26], esrc, edst, ew, ewsum,
          NE, GB, g1);
  // --- gat1b: g1 (256) -> 64, mean heads -> A2 ---
  run_gat(stream, g1, 256, 256, 1, 64, 0, 0, in[31], in[27], in[28], in[29],
          in[30], in[32], esrc, edst, ew, ewsum, NE, GB, A2);

  // --- MHCA against latent arrays ---
  run_mhca(stream, A1, in[3], in[10], in[9], in[6], in[5], in[12], in[11],
           in[8], in[7], qbuf, kbuf, vbuf, attn, B1, GB.apack, GB.bpack);
  run_mhca(stream, A2, in[4], in[18], in[17], in[14], in[13], in[20], in[19],
           in[16], in[15], qbuf, kbuf, vbuf, attn, B2, GB.apack, GB.bpack);

  // --- concat -> gat2a (elu) -> gat2b -> output ---
  concat_kernel<<<(NS * 128 + 255) / 256, 256, 0, stream>>>(B1, B2, cbuf, NS);
  run_gat(stream, cbuf, 128, 128, 4, 64, 1, 1, in[37], in[33], in[34], in[35],
          in[36], in[38], esrc, edst, ew, ewsum, NE, GB, g2);
  run_gat(stream, g2, 256, 256, 1, 1, 0, 0, in[43], in[39], in[40], in[41],
          in[42], in[44], esrc, edst, ew, ewsum, NE, GB, (float*)d_out);
}
